// SparseConv3d_InPlace_Autograd_88373247082551
// MI455X (gfx1250) — compile-verified, matrix operand order checked by reading, not run on hardware
//
#include <hip/hip_runtime.h>

typedef __attribute__((ext_vector_type(16))) __bf16 v16bf;
typedef __attribute__((ext_vector_type(8)))  float  v8f;

#define NPTS   200000
#define CIN    64
#define COUT   64
#define K3     27
#define RULES  100000
#define TILE_R 128   // rules per workgroup: 8 waves x 16 rules

// out[i] = bias[i % 64]  (reference adds bias to every row, scattered or not)
__global__ __launch_bounds__(256) void init_out_kernel(float* __restrict__ out,
                                                       const float* __restrict__ bias,
                                                       int total) {
  int i = blockIdx.x * blockDim.x + threadIdx.x;
  if (i < total) out[i] = bias[i & (COUT - 1)];
}

__global__ __launch_bounds__(256) void spconv_kernel(const float* __restrict__ x,
                                                     const int*   __restrict__ in_inds,
                                                     const int*   __restrict__ out_inds,
                                                     const float* __restrict__ w,
                                                     float*       __restrict__ out) {
  const int k    = blockIdx.y;              // kernel offset 0..26
  const int r0   = blockIdx.x * TILE_R;     // first rule of this workgroup
  const int tid  = threadIdx.x;
  const int wave = tid >> 5;
  const int lane = tid & 31;
  const int lrow = lane & 15;               // M (for A) or N (for B and C/D) within 16x16 tile
  const int g    = lane >> 4;               // lane half-group

  // ---------------------------------------------------------------
  // Stage B fragments into LDS: weights[k] (64x64 f32) split into
  // bf16 hi/lo, pre-swizzled into the exact per-lane WMMA B layout.
  // 16 fragments: [nt(4)][ks(2)][part(2)], each = 32 lanes x 16 bf16.
  // e == (((nt*2+ks)*2+part)*32 + lane_)*16 + h by bit construction.
  // ---------------------------------------------------------------
  __shared__ __align__(32) __bf16 Bfrag[16 * 32 * 16];
  for (int e = tid; e < 16 * 32 * 16; e += 256) {
    int h     = e & 15;
    int lane_ = (e >> 4) & 31;
    int part  = (e >> 9) & 1;
    int ks    = (e >> 10) & 1;
    int nt    = (e >> 11) & 3;
    int g_    = lane_ >> 4;
    int n     = nt * 16 + (lane_ & 15);                       // Cout column
    int kk    = ks * 32 + ((h < 8) ? h : (h + 8)) + g_ * 8;   // Cin row (K)
    float wv  = w[((size_t)k * CIN + kk) * COUT + n];
    __bf16 hi = (__bf16)wv;
    Bfrag[e]  = (part == 0) ? hi : (__bf16)(wv - (float)hi);
  }
  __syncthreads();

  // ---------------------------------------------------------------
  // Gather A rows (16 rules per wave) straight into registers.
  // 16-bit A layout: lane group g needs K = base+[0..7] and base+[16..23]
  // with base = ks*32 + g*8 -> two contiguous 8-float runs per slice.
  // ---------------------------------------------------------------
  const int  rule_in = r0 + wave * 16 + lrow;
  const bool vin     = rule_in < RULES;
  const int  iidx    = vin ? in_inds[k * RULES + rule_in] : 0;
  const float* xr    = x + (size_t)iidx * CIN;

  v16bf Ahi[2], Alo[2];
#pragma unroll
  for (int ks = 0; ks < 2; ++ks) {
    int base  = ks * 32 + g * 8;
    float4 a0 = *(const float4*)(xr + base + 0);
    float4 a1 = *(const float4*)(xr + base + 4);
    float4 a2 = *(const float4*)(xr + base + 16);
    float4 a3 = *(const float4*)(xr + base + 20);
    float vals[16] = {a0.x, a0.y, a0.z, a0.w, a1.x, a1.y, a1.z, a1.w,
                      a2.x, a2.y, a2.z, a2.w, a3.x, a3.y, a3.z, a3.w};
#pragma unroll
    for (int h = 0; h < 16; ++h) {
      float f   = vin ? vals[h] : 0.0f;
      __bf16 hi = (__bf16)f;
      Ahi[ks][h] = hi;
      Alo[ks][h] = (__bf16)(f - (float)hi);
    }
  }

  // Output rows for this wave's tile: C/D layout -> lane group g holds M = 8*g + j.
  int  orows[8];
  bool oval[8];
#pragma unroll
  for (int j = 0; j < 8; ++j) {
    int ro   = r0 + wave * 16 + 8 * g + j;
    oval[j]  = ro < RULES;
    orows[j] = oval[j] ? out_inds[k * RULES + ro] : 0;
  }

  // ---------------------------------------------------------------
  // 4 N-tiles x (2 K-slices x 3 error-compensated bf16 WMMAs) each,
  // then scatter-add the 16x16 f32 tile with native f32 atomics.
  // ---------------------------------------------------------------
  const v16bf* Bv = (const v16bf*)Bfrag;
#pragma unroll
  for (int nt = 0; nt < 4; ++nt) {
    v8f acc = {};
#pragma unroll
    for (int ks = 0; ks < 2; ++ks) {
      v16bf bhi = Bv[((nt * 2 + ks) * 2 + 0) * 32 + lane];
      v16bf blo = Bv[((nt * 2 + ks) * 2 + 1) * 32 + lane];
      acc = __builtin_amdgcn_wmma_f32_16x16x32_bf16(false, Ahi[ks], false, bhi,
                                                    (short)0, acc, false, false);
      acc = __builtin_amdgcn_wmma_f32_16x16x32_bf16(false, Alo[ks], false, bhi,
                                                    (short)0, acc, false, false);
      acc = __builtin_amdgcn_wmma_f32_16x16x32_bf16(false, Ahi[ks], false, blo,
                                                    (short)0, acc, false, false);
    }
    int ncol = nt * 16 + lrow;
#pragma unroll
    for (int j = 0; j < 8; ++j) {
      if (oval[j]) {
        float* p = out + (size_t)orows[j] * COUT + ncol;
        __hip_atomic_fetch_add(p, acc[j], __ATOMIC_RELAXED, __HIP_MEMORY_SCOPE_AGENT);
      }
    }
  }
}

extern "C" void kernel_launch(void* const* d_in, const int* in_sizes, int n_in,
                              void* d_out, int out_size, void* d_ws, size_t ws_size,
                              hipStream_t stream) {
  const float* x        = (const float*)d_in[0];
  const int*   in_inds  = (const int*)d_in[1];
  const int*   out_inds = (const int*)d_in[2];
  const float* w        = (const float*)d_in[3];
  const float* bias     = (const float*)d_in[4];
  float*       out      = (float*)d_out;

  const int total = NPTS * COUT;
  init_out_kernel<<<(total + 255) / 256, 256, 0, stream>>>(out, bias, total);

  dim3 grid((RULES + TILE_R - 1) / TILE_R, K3);
  spconv_kernel<<<grid, 256, 0, stream>>>(x, in_inds, out_inds, w, out);
}
